// OptimizedClassifyModule_47270410059924
// MI455X (gfx1250) — compile-verified, matrix operand order checked by reading, not run on hardware
//
#include <hip/hip_runtime.h>

// ---------------------------------------------------------------------------
// out[b,c] = sum_n (|x|>eps && mask!=0 ? x : 0) * w[c,n] + bias[c]
// B=32, C=2, N=96^3=884736.  Skinny GEMM (32xN)@(Nx2), memory bound:
// ~124 MB HBM traffic -> ~5.3us floor @ 23.3 TB/s.  WMMA f16 (f32 accum)
// used as the reduction engine; coalesced float4 streaming of x.
// ---------------------------------------------------------------------------

typedef _Float16 v4h  __attribute__((ext_vector_type(4)));
typedef _Float16 v8h  __attribute__((ext_vector_type(8)));
typedef _Float16 v16h __attribute__((ext_vector_type(16)));
typedef float    v8f  __attribute__((ext_vector_type(8)));

#define NSPAT   884736   // 96^3
#define CHUNK   8192     // K elements per block
#define NCHUNKS 108      // NSPAT / CHUNK
#define KTILE   1024     // K elements per LDS stage
#define STAGES  8        // CHUNK / KTILE
#define THREADS 256
#define NWAVES  8        // THREADS / 32
#define KSLICE  128      // KTILE / NWAVES (per-wave K ownership)
#define KSTRIDE 1032     // KTILE + 8 halves pad (keeps 16B alignment, skews banks)
#define EPSV    1e-8f

__global__ __launch_bounds__(THREADS)
void cls_wmma_kernel(const float* __restrict__ x,
                     const int*   __restrict__ mask,
                     const float* __restrict__ w,
                     float*       __restrict__ ws)
{
    __shared__ __align__(16) _Float16 lx[16][KSTRIDE]; // gated x, f16
    __shared__ __align__(16) _Float16 lw[2][KSTRIDE];  // weights, f16
    __shared__ float red[NWAVES][2][16];

    const int t     = threadIdx.x;
    const int wave  = t >> 5;
    const int lane  = t & 31;
    const int bg    = blockIdx.y;                 // batch group: batches [bg*16, bg*16+16)
    const int chunk = blockIdx.x;
    const size_t kchunk = (size_t)chunk * CHUNK;

    v8f acc = {};                                  // 16x16 f32 C/D tile (cols 0..1 real)

    const int kt = t * 4;                          // this thread's K offset in the tile

    for (int stage = 0; stage < STAGES; ++stage) {
        const size_t k0 = kchunk + (size_t)stage * KTILE;
        __syncthreads();                           // LDS reuse guard

        // ---- stage: mask + weights + gated x into LDS (coalesced b128 loads)
        const int4   mk = *(const int4*)  (mask + k0 + kt);
        const float4 w0 = *(const float4*)(w          + k0 + kt);
        const float4 w1 = *(const float4*)(w + NSPAT  + k0 + kt);
        {
            v4h h0 = { (_Float16)w0.x, (_Float16)w0.y, (_Float16)w0.z, (_Float16)w0.w };
            v4h h1 = { (_Float16)w1.x, (_Float16)w1.y, (_Float16)w1.z, (_Float16)w1.w };
            *(v4h*)&lw[0][kt] = h0;
            *(v4h*)&lw[1][kt] = h1;
        }
        #pragma unroll
        for (int m = 0; m < 16; ++m) {
            const float4 xv = *(const float4*)(x + (size_t)(bg * 16 + m) * NSPAT + k0 + kt);
            const float gx = (fabsf(xv.x) > EPSV && mk.x != 0) ? xv.x : 0.0f;
            const float gy = (fabsf(xv.y) > EPSV && mk.y != 0) ? xv.y : 0.0f;
            const float gz = (fabsf(xv.z) > EPSV && mk.z != 0) ? xv.z : 0.0f;
            const float gw = (fabsf(xv.w) > EPSV && mk.w != 0) ? xv.w : 0.0f;
            v4h g = { (_Float16)gx, (_Float16)gy, (_Float16)gz, (_Float16)gw };
            *(v4h*)&lx[m][kt] = g;
        }
        __syncthreads();

        // ---- compute: each wave owns K slice [wave*128, wave*128+128) -> 4 WMMAs
        const int kw   = wave * KSLICE;
        const int mrow = lane & 15;
        const int b1   = (lane < 16) ? 0 : 8;      // 16-bit A 16x32 lane layout
        #pragma unroll
        for (int s = 0; s < 4; ++s) {
            const int kk = kw + s * 32;
            const v8h a_lo = *(const v8h*)&lx[mrow][kk + b1];
            const v8h a_hi = *(const v8h*)&lx[mrow][kk + b1 + 16];
            const v16h A = __builtin_shufflevector(a_lo, a_hi,
                0,1,2,3,4,5,6,7,8,9,10,11,12,13,14,15);
            v8h bl = {}, bh = {};
            if (mrow < 2) {                         // only 2 real weight columns
                bl = *(const v8h*)&lw[mrow][kk + b1];
                bh = *(const v8h*)&lw[mrow][kk + b1 + 16];
            }
            const v16h B = __builtin_shufflevector(bl, bh,
                0,1,2,3,4,5,6,7,8,9,10,11,12,13,14,15);
            // D = A(16x32 f16) x B(32x16 f16) + C(f32)
            acc = __builtin_amdgcn_wmma_f32_16x16x32_f16(
                false, A, false, B, (short)0, acc, false, false);
        }
    }

    // ---- cross-wave reduction of the two live columns of the C tile
    {
        const int n = lane & 15;                    // column index
        if (n < 2) {
            const int mb = (lane < 16) ? 0 : 8;     // C/D layout: VGPR r -> M = r (+8)
            #pragma unroll
            for (int r = 0; r < 8; ++r) red[wave][n][mb + r] = acc[r];
        }
    }
    __syncthreads();
    if (t < 32) {
        const int c = t >> 4, m = t & 15;
        float s = 0.0f;
        #pragma unroll
        for (int wv = 0; wv < NWAVES; ++wv) s += red[wv][c][m];
        ws[(((size_t)bg * NCHUNKS + chunk) * 16 + m) * 2 + c] = s;
    }
}

__global__ void cls_finalize(const float* __restrict__ ws,
                             const float* __restrict__ bias,
                             float*       __restrict__ out)
{
    const int t = threadIdx.x;
    if (t >= 64) return;
    const int b = t >> 1, c = t & 1;
    const int bg = b >> 4, m = b & 15;
    float s = 0.0f;
    for (int ch = 0; ch < NCHUNKS; ++ch)
        s += ws[(((size_t)bg * NCHUNKS + ch) * 16 + m) * 2 + c];
    out[b * 2 + c] = s + bias[c];
}

extern "C" void kernel_launch(void* const* d_in, const int* in_sizes, int n_in,
                              void* d_out, int out_size, void* d_ws, size_t ws_size,
                              hipStream_t stream) {
    const float* x    = (const float*)d_in[0];   // (32,1,96,96,96) f32
    const int*   mask = (const int*)  d_in[1];   // (1,1,96,96,96) i32
    const float* w    = (const float*)d_in[2];   // (2,1,96,96,96) f32
    const float* bias = (const float*)d_in[3];   // (2,) f32
    float* out = (float*)d_out;                  // (32,2) f32
    float* ws  = (float*)d_ws;                   // 2*108*16*2 floats = 27.6 KB

    dim3 grid(NCHUNKS, 2);
    cls_wmma_kernel<<<grid, THREADS, 0, stream>>>(x, mask, w, ws);
    cls_finalize<<<1, 64, 0, stream>>>(ws, bias, out);
}
